// LocalCorrelationMatcher_20040317403410
// MI455X (gfx1250) — matrix operand order repaired and checked by reading, not compiled, not run on hardware
//
#include <hip/hip_runtime.h>
#include <hip/hip_bf16.h>
#include <math.h>

typedef __attribute__((ext_vector_type(16))) _Float16 v16h;
typedef __attribute__((ext_vector_type(8)))  float    v8f;
typedef int vint4 __attribute__((vector_size(16)));
typedef _Float16 f16;

namespace {
constexpr int B_    = 4;
constexpr int C_    = 256;     // feature channels
constexpr int H_    = 64;
constexpr int W_    = 128;
constexpr int HID_  = 128;
constexpr int CIN1_ = 608;     // 596 real concat channels padded to 19*32
constexpr int NPIX_ = H_ * W_; // 8192
constexpr int TOT_  = B_ * NPIX_; // 32768
}

// CDNA5 async global->LDS staging (ASYNCcnt) with synchronous fallback.
#if __has_builtin(__builtin_amdgcn_global_load_async_to_lds_b128)
#define USE_ASYNC_LDS 1
#else
#define USE_ASYNC_LDS 0
#endif

#if USE_ASYNC_LDS
// arg0 type per compiler diagnostic: addrspace(1) int4* ; arg1: addrspace(3)
#define ASYNC_GLB(p) ((__attribute__((address_space(1))) vint4*)((const void*)(p)))
#define ASYNC_LDS(p) ((__attribute__((address_space(3))) vint4*)((void*)(p)))
#endif

__device__ __forceinline__ void wait_async_lds() {
#if USE_ASYNC_LDS
#if __has_builtin(__builtin_amdgcn_s_wait_asynccnt)
  __builtin_amdgcn_s_wait_asynccnt(0);
#else
  asm volatile("s_wait_asynccnt 0x0" ::: "memory");
#endif
#endif
}

// ---------------------------------------------------------------------------
// Weight repack: w1 (128,596,3,3) f32 -> Wt1[t][oc][c] f16, c padded to 608
// ---------------------------------------------------------------------------
__global__ void k_prep_w1(const float* __restrict__ w1, f16* __restrict__ wt) {
  int i = blockIdx.x * 256 + threadIdx.x;
  if (i >= 9 * HID_ * CIN1_) return;
  int c  = i % CIN1_;
  int oc = (i / CIN1_) % HID_;
  int t  = i / (CIN1_ * HID_);
  float v = 0.0f;
  if (c < 596) v = w1[(oc * 596 + c) * 9 + t];
  wt[((size_t)t * HID_ + oc) * CIN1_ + c] = (f16)v;
}

__global__ void k_prep_w2(const float* __restrict__ w2, f16* __restrict__ wt) {
  int i = blockIdx.x * 256 + threadIdx.x;
  if (i >= 9 * HID_ * HID_) return;
  int c  = i % HID_;
  int oc = (i / HID_) % HID_;
  int t  = i / (HID_ * HID_);
  wt[((size_t)t * HID_ + oc) * HID_ + c] = (f16)w2[(oc * HID_ + c) * 9 + t];
}

// ---------------------------------------------------------------------------
// Warp feat_b by prior_flow (bilinear, zero outside), write fa/fbw/flow/conf
// into the padded f16 concat tensor x, and emit L2-normalized fields for corr.
// ---------------------------------------------------------------------------
__global__ void k_build(const float* __restrict__ fa, const float* __restrict__ fb,
                        const float* __restrict__ flow, const float* __restrict__ conf,
                        f16* __restrict__ xb, f16* __restrict__ an, f16* __restrict__ bn) {
  int p = blockIdx.x * 256 + threadIdx.x;
  if (p >= TOT_) return;
  int b = p / NPIX_;
  int hw = p % NPIX_;
  int h = hw / W_, w = hw % W_;

  float fx = flow[((size_t)(b * 2 + 0) * H_ + h) * W_ + w];
  float fy = flow[((size_t)(b * 2 + 1) * H_ + h) * W_ + w];
  float px = (float)w + fx, py = (float)h + fy;
  float x0f = floorf(px), y0f = floorf(py);
  float wx = px - x0f, wy = py - y0f;
  int x0 = (int)x0f, y0 = (int)y0f;
  int x1 = x0 + 1, y1 = y0 + 1;

  auto cl = [](int v, int lo, int hi) { return v < lo ? lo : (v > hi ? hi : v); };
  float m00 = ((x0 >= 0 && x0 < W_ && y0 >= 0 && y0 < H_) ? 1.f : 0.f) * (1.f - wx) * (1.f - wy);
  float m01 = ((x1 >= 0 && x1 < W_ && y0 >= 0 && y0 < H_) ? 1.f : 0.f) * wx * (1.f - wy);
  float m10 = ((x0 >= 0 && x0 < W_ && y1 >= 0 && y1 < H_) ? 1.f : 0.f) * (1.f - wx) * wy;
  float m11 = ((x1 >= 0 && x1 < W_ && y1 >= 0 && y1 < H_) ? 1.f : 0.f) * wx * wy;
  int cx0 = cl(x0, 0, W_ - 1), cx1 = cl(x1, 0, W_ - 1);
  int cy0 = cl(y0, 0, H_ - 1), cy1 = cl(y1, 0, H_ - 1);
  int i00 = cy0 * W_ + cx0, i01 = cy0 * W_ + cx1;
  int i10 = cy1 * W_ + cx0, i11 = cy1 * W_ + cx1;

  float suma = 0.f, sumb = 0.f;
  for (int c = 0; c < C_; ++c) {
    size_t base = (size_t)(b * C_ + c) * NPIX_;
    float a = fa[base + hw];
    float v = m00 * fb[base + i00] + m01 * fb[base + i01] +
              m10 * fb[base + i10] + m11 * fb[base + i11];
    suma += a * a;
    sumb += v * v;
    xb[((size_t)(b * CIN1_ + c) * NPIX_) + hw]       = (f16)a;
    xb[((size_t)(b * CIN1_ + 256 + c) * NPIX_) + hw] = (f16)v;
  }
  float inva = 1.f / fmaxf(sqrtf(suma), 1e-12f);
  float invb = 1.f / fmaxf(sqrtf(sumb), 1e-12f);
  for (int c = 0; c < C_; ++c) {
    float a = (float)xb[((size_t)(b * CIN1_ + c) * NPIX_) + hw];
    float v = (float)xb[((size_t)(b * CIN1_ + 256 + c) * NPIX_) + hw];
    an[((size_t)(b * C_ + c) * NPIX_) + hw] = (f16)(a * inva);
    bn[((size_t)(b * C_ + c) * NPIX_) + hw] = (f16)(v * invb);
  }
  xb[((size_t)(b * CIN1_ + 512) * NPIX_) + hw] = (f16)fx;
  xb[((size_t)(b * CIN1_ + 513) * NPIX_) + hw] = (f16)fy;
  xb[((size_t)(b * CIN1_ + 514) * NPIX_) + hw] = (f16)conf[(size_t)b * NPIX_ + hw];
  for (int c = 596; c < CIN1_; ++c)
    xb[((size_t)(b * CIN1_ + c) * NPIX_) + hw] = (f16)0.f;
}

// ---------------------------------------------------------------------------
// 81-offset local correlation of normalized fields -> x channels 515..595
// ---------------------------------------------------------------------------
__global__ void k_corr(const f16* __restrict__ an, const f16* __restrict__ bn,
                       f16* __restrict__ xb) {
  int p = blockIdx.x * 256 + threadIdx.x;
  if (p >= TOT_) return;
  int b = p / NPIX_;
  int hw = p % NPIX_;
  int h = hw / W_, w = hw % W_;

  for (int dy = -4; dy <= 4; ++dy) {
    int hh = h + dy;
    float acc[9];
#pragma unroll
    for (int i = 0; i < 9; ++i) acc[i] = 0.f;
    if (hh >= 0 && hh < H_) {
      for (int c = 0; c < C_; ++c) {
        float a = (float)an[((size_t)(b * C_ + c) * NPIX_) + hw];
        const f16* row = bn + (size_t)(b * C_ + c) * NPIX_ + hh * W_;
#pragma unroll
        for (int dx = 0; dx < 9; ++dx) {
          int ww = w + dx - 4;
          float bv = (ww >= 0 && ww < W_) ? (float)row[ww] : 0.f;
          acc[dx] += a * bv;
        }
      }
    }
    int chbase = 515 + (dy + 4) * 9;
    for (int dx = 0; dx < 9; ++dx)
      xb[((size_t)(b * CIN1_ + chbase + dx) * NPIX_) + hw] = (f16)acc[dx];
  }
}

// ---------------------------------------------------------------------------
// Implicit-GEMM 3x3 conv via v_wmma_f32_16x16x32_f16, software-pipelined.
// One block = one image row (128 pixels) x all 128 output channels.
// 8 waves: mg=wv>>1 (4 M-groups of 32 px), ng=wv&1 (2 N-groups of 64 oc);
// each wave holds 2x4 WMMA f32 accumulators.
// Double-buffered LDS; B (weight) tile staged via async global->LDS b128
// (ASYNCcnt) when available, A (im2col) tile via register pipeline since it
// needs transpose + zero-fill. One barrier per K-step.
// LDS rows are 40 halves (80B, 16B aligned) so fragments load as ds_load_b128.
// ---------------------------------------------------------------------------
__global__ __launch_bounds__(256) void k_conv_wmma(
    const f16* __restrict__ xp, const f16* __restrict__ wt,
    const float* __restrict__ bias, float* __restrict__ y, int cinp) {
  __shared__ f16 As[2][128 * 40];
  __shared__ f16 Bs[2][128 * 40];

  int tid = threadIdx.x;
  int lane = tid & 31;
  int wv = tid >> 5;
  int b = blockIdx.x / H_;
  int h = blockIdx.x % H_;
  int M0 = (wv >> 1) * 32;
  int N0 = (wv & 1) * 64;

  const int nck = cinp >> 5;     // K chunks of 32 per tap
  const int nsteps = 9 * nck;

  v8f acc[2][4];
  v8f z = {0.f, 0.f, 0.f, 0.f, 0.f, 0.f, 0.f, 0.f};
#pragma unroll
  for (int i = 0; i < 2; ++i)
#pragma unroll
    for (int j = 0; j < 4; ++j) acc[i][j] = z;

  // ---- A tile: global -> registers (transpose + zero-fill handled here)
  f16 aregs[16];
  auto fetchA = [&](int step) {
    int t = step / nck, ck = step - t * nck;
    int dy = t / 3 - 1, dx = t % 3 - 1;
    int hh = h + dy;
    bool hok = (hh >= 0 && hh < H_);
    int c0 = ck << 5;
    int k = tid >> 3, wseg = tid & 7;
    const f16* xrow =
        xp + (size_t)(b * cinp + c0 + k) * NPIX_ + (hok ? hh * W_ : 0);
#pragma unroll
    for (int j = 0; j < 16; ++j) {
      int wpix = wseg * 16 + j;
      int ww = wpix + dx;
      f16 v = (f16)0.f;
      if (hok && ww >= 0 && ww < W_) v = xrow[ww];
      aregs[j] = v;
    }
  };
  auto storeA = [&](int buf) {
    int k = tid >> 3, wseg = tid & 7;
#pragma unroll
    for (int j = 0; j < 16; ++j) As[buf][(wseg * 16 + j) * 40 + k] = aregs[j];
  };

  // ---- B tile: Bs[oc][k] = Wt[t][oc][c0+k] (contiguous b128 copies)
  auto stageB = [&](int step, int buf) {
    int t = step / nck, ck = step - t * nck;
    int c0 = ck << 5;
    int oc = tid >> 1, part = tid & 1;
    const f16* src = wt + ((size_t)t * HID_ + oc) * cinp + c0 + part * 16;
    f16* dst = &Bs[buf][oc * 40 + part * 16];
#if USE_ASYNC_LDS
    __builtin_amdgcn_global_load_async_to_lds_b128(ASYNC_GLB(src),
                                                   ASYNC_LDS(dst), 0, 0);
    __builtin_amdgcn_global_load_async_to_lds_b128(ASYNC_GLB(src + 8),
                                                   ASYNC_LDS(dst + 8), 0, 0);
#else
    const uint4* s = reinterpret_cast<const uint4*>(src);
    uint4* d = reinterpret_cast<uint4*>(dst);
    d[0] = s[0];
    d[1] = s[1];
#endif
  };

  // ---- prologue: stage step 0 into buffer 0
  fetchA(0);
  stageB(0, 0);
  storeA(0);
  wait_async_lds();
  __syncthreads();

  for (int step = 0; step < nsteps; ++step) {
    int cur = step & 1;
    int nxt = cur ^ 1;
    bool more = (step + 1 < nsteps);
    if (more) {
      fetchA(step + 1);       // global loads overlap the WMMAs below
      stageB(step + 1, nxt);  // async copies land in the other buffer
    }

    // ---- fragments in contiguous k order (k = 16*(lane/16) + s), same for A and B
    v16h afrag[2], bfrag[4];
    int klo = (lane < 16) ? 0 : 16;  // A: lanes 0-15 K 0..15, 16-31 K 16..31
#pragma unroll
    for (int mi = 0; mi < 2; ++mi) {
      int row = M0 + mi * 16 + (lane & 15);
      union { uint4 u[2]; v16h v; } tmp;
      tmp.u[0] = *reinterpret_cast<const uint4*>(&As[cur][row * 40 + klo]);
      tmp.u[1] = *reinterpret_cast<const uint4*>(&As[cur][row * 40 + klo + 8]);
      afrag[mi] = tmp.v;
    }
    int koff = (lane < 16) ? 0 : 16; // B: lanes 0-15 K 0..15, 16-31 K 16..31
#pragma unroll
    for (int ni = 0; ni < 4; ++ni) {
      int col = N0 + ni * 16 + (lane & 15);
      union { uint4 u[2]; v16h v; } tmp;
      const uint4* bp = reinterpret_cast<const uint4*>(&Bs[cur][col * 40 + koff]);
      tmp.u[0] = bp[0];
      tmp.u[1] = bp[1];
      bfrag[ni] = tmp.v;
    }

#pragma unroll
    for (int mi = 0; mi < 2; ++mi)
#pragma unroll
      for (int ni = 0; ni < 4; ++ni)
        acc[mi][ni] = __builtin_amdgcn_wmma_f32_16x16x32_f16(
            false, afrag[mi], false, bfrag[ni], (short)0, acc[mi][ni],
            false, false);

    if (more) {
      storeA(nxt);
      wait_async_lds();
    }
    __syncthreads();
  }

  // ---- store: C/D layout -> lanes 0-15 rows 0..7, lanes 16-31 rows 8..15.
  // Each lane's 8 rows are 8 consecutive pixels -> two b128 stores per tile.
  int nloc = lane & 15;
  int mb = (lane >> 4) * 8;
#pragma unroll
  for (int mi = 0; mi < 2; ++mi)
#pragma unroll
    for (int ni = 0; ni < 4; ++ni) {
      int n = N0 + ni * 16 + nloc;
      float bv = bias[n];
      int m = M0 + mi * 16 + mb;
      float* dst = y + ((size_t)(b * HID_ + n)) * NPIX_ + h * W_ + m;
      float4 lo, hi;
      lo.x = acc[mi][ni][0] + bv;
      lo.y = acc[mi][ni][1] + bv;
      lo.z = acc[mi][ni][2] + bv;
      lo.w = acc[mi][ni][3] + bv;
      hi.x = acc[mi][ni][4] + bv;
      hi.y = acc[mi][ni][5] + bv;
      hi.z = acc[mi][ni][6] + bv;
      hi.w = acc[mi][ni][7] + bv;
      reinterpret_cast<float4*>(dst)[0] = lo;
      reinterpret_cast<float4*>(dst)[1] = hi;
    }
}

// ---------------------------------------------------------------------------
// GroupNorm stats: one block per (batch, group); 16 channels x 8192 pixels
// ---------------------------------------------------------------------------
__global__ void k_gnstats(const float* __restrict__ y, float* __restrict__ stats) {
  int bg = blockIdx.x;
  int b = bg >> 3, g = bg & 7;
  int cbase = g * 16;
  float s = 0.f, ss = 0.f;
  for (int i = threadIdx.x; i < 16 * NPIX_; i += 256) {
    int c = cbase + (i >> 13);
    int pix = i & (NPIX_ - 1);
    float v = y[((size_t)(b * HID_ + c)) * NPIX_ + pix];
    s += v;
    ss += v * v;
  }
  __shared__ float rs[256], rss[256];
  rs[threadIdx.x] = s;
  rss[threadIdx.x] = ss;
  __syncthreads();
  for (int o = 128; o > 0; o >>= 1) {
    if (threadIdx.x < o) {
      rs[threadIdx.x] += rs[threadIdx.x + o];
      rss[threadIdx.x] += rss[threadIdx.x + o];
    }
    __syncthreads();
  }
  if (threadIdx.x == 0) {
    float n = 16.f * (float)NPIX_;
    float m = rs[0] / n;
    float var = rss[0] / n - m * m;
    stats[bg * 2 + 0] = m;
    stats[bg * 2 + 1] = rsqrtf(var + 1e-5f);
  }
}

// ---------------------------------------------------------------------------
// GroupNorm apply + exact GELU -> f16 activations for next conv
// ---------------------------------------------------------------------------
__global__ void k_gnapply(const float* __restrict__ y, const float* __restrict__ stats,
                          const float* __restrict__ sc, const float* __restrict__ bi,
                          f16* __restrict__ hout) {
  int i = blockIdx.x * 256 + threadIdx.x;
  if (i >= B_ * HID_ * NPIX_) return;
  int c = (i / NPIX_) % HID_;
  int b = i / (HID_ * NPIX_);
  int g = c >> 4;
  float m = stats[(b * 8 + g) * 2 + 0];
  float r = stats[(b * 8 + g) * 2 + 1];
  float v = (y[i] - m) * r * sc[c] + bi[c];
  float gel = 0.5f * v * (1.f + erff(v * 0.70710678118654752f));
  hout[i] = (f16)gel;
}

// ---------------------------------------------------------------------------
// Output heads: 3x3 convs to 2 (tanh) + 1 (sigmoid) channels
// ---------------------------------------------------------------------------
__global__ void k_heads(const f16* __restrict__ h2, const float* __restrict__ wd,
                        const float* __restrict__ bd, const float* __restrict__ wc,
                        const float* __restrict__ bc, float* __restrict__ out) {
  int p = blockIdx.x * 256 + threadIdx.x;
  if (p >= TOT_) return;
  int b = p / NPIX_;
  int hw = p % NPIX_;
  int h = hw / W_, w = hw % W_;
  float a0 = 0.f, a1 = 0.f, a2 = 0.f;
  for (int t = 0; t < 9; ++t) {
    int dy = t / 3 - 1, dx = t % 3 - 1;
    int hh = h + dy, ww = w + dx;
    if (hh < 0 || hh >= H_ || ww < 0 || ww >= W_) continue;
    for (int c = 0; c < HID_; ++c) {
      float v = (float)h2[((size_t)(b * HID_ + c)) * NPIX_ + hh * W_ + ww];
      a0 += v * wd[(0 * HID_ + c) * 9 + t];
      a1 += v * wd[(1 * HID_ + c) * 9 + t];
      a2 += v * wc[c * 9 + t];
    }
  }
  out[((size_t)(b * 2 + 0)) * NPIX_ + hw] = tanhf(a0 + bd[0]);
  out[((size_t)(b * 2 + 1)) * NPIX_ + hw] = tanhf(a1 + bd[1]);
  out[(size_t)B_ * 2 * NPIX_ + (size_t)b * NPIX_ + hw] =
      1.f / (1.f + expf(-(a2 + bc[0])));
}

// ---------------------------------------------------------------------------
extern "C" void kernel_launch(void* const* d_in, const int* in_sizes, int n_in,
                              void* d_out, int out_size, void* d_ws, size_t ws_size,
                              hipStream_t stream) {
  (void)in_sizes; (void)n_in; (void)out_size; (void)ws_size;
  const float* fa   = (const float*)d_in[0];
  const float* fb   = (const float*)d_in[1];
  const float* flow = (const float*)d_in[2];
  const float* conf = (const float*)d_in[3];
  const float* w1   = (const float*)d_in[4];
  const float* b1   = (const float*)d_in[5];
  const float* g1s  = (const float*)d_in[6];
  const float* g1b  = (const float*)d_in[7];
  const float* w2   = (const float*)d_in[8];
  const float* b2   = (const float*)d_in[9];
  const float* g2s  = (const float*)d_in[10];
  const float* g2b  = (const float*)d_in[11];
  const float* wd   = (const float*)d_in[12];
  const float* bd   = (const float*)d_in[13];
  const float* wc   = (const float*)d_in[14];
  const float* bc   = (const float*)d_in[15];
  float* out = (float*)d_out;

  // workspace layout (bytes), with lifetime-based aliasing:
  //  [0, 39845888)            x f16  (B x 608 x H x W)         [build..conv1]
  //  [39845888, +16777216)    anorm f16 | later y f32 (conv out)
  //  [56623104, +16777216)    bnorm f16 | later h f16 (activations)
  //  [73400320, +1400832)     Wt1 f16
  //  [74801152, +294912)      Wt2 f16
  //  [75096064, +256)         GN stats
  char* ws = (char*)d_ws;
  f16*   xb    = (f16*)(ws + 0);
  f16*   an    = (f16*)(ws + 39845888ull);
  f16*   bn    = (f16*)(ws + 56623104ull);
  float* ybuf  = (float*)(ws + 39845888ull);  // aliases an (dead after corr)
  f16*   hbuf  = (f16*)(ws + 56623104ull);    // aliases bn (dead after corr)
  f16*   wt1   = (f16*)(ws + 73400320ull);
  f16*   wt2   = (f16*)(ws + 74801152ull);
  float* stats = (float*)(ws + 75096064ull);

  k_prep_w1<<<(9 * HID_ * CIN1_ + 255) / 256, 256, 0, stream>>>(w1, wt1);
  k_prep_w2<<<(9 * HID_ * HID_ + 255) / 256, 256, 0, stream>>>(w2, wt2);
  k_build<<<TOT_ / 256, 256, 0, stream>>>(fa, fb, flow, conf, xb, an, bn);
  k_corr<<<TOT_ / 256, 256, 0, stream>>>(an, bn, xb);

  // conv1: 596(+pad)->128, WMMA implicit GEMM
  k_conv_wmma<<<B_ * H_, 256, 0, stream>>>(xb, wt1, b1, ybuf, CIN1_);
  k_gnstats<<<32, 256, 0, stream>>>(ybuf, stats);
  k_gnapply<<<(B_ * HID_ * NPIX_) / 256, 256, 0, stream>>>(ybuf, stats, g1s, g1b, hbuf);

  // conv2: 128->128
  k_conv_wmma<<<B_ * H_, 256, 0, stream>>>(hbuf, wt2, b2, ybuf, HID_);
  k_gnstats<<<32, 256, 0, stream>>>(ybuf, stats);
  k_gnapply<<<(B_ * HID_ * NPIX_) / 256, 256, 0, stream>>>(ybuf, stats, g2s, g2b, hbuf);

  k_heads<<<TOT_ / 256, 256, 0, stream>>>(hbuf, wd, bd, wc, bc, out);
}
